// SoftPointerNetwork_6554120093863
// MI455X (gfx1250) — matrix operand order repaired and checked by reading, not compiled, not run on hardware
//
#include <hip/hip_runtime.h>

// ---------------------------------------------------------------------------
// SoftPointerNetwork for MI455X (gfx1250, wave32, WMMA bf16 16x16x32)
// ---------------------------------------------------------------------------

typedef __attribute__((ext_vector_type(16))) __bf16 bf16x16;
typedef __attribute__((ext_vector_type(8)))  float  f32x8;

static constexpr int kB   = 16;
static constexpr int kT   = 128;
static constexpr int kS   = 1024;
static constexpr int kH   = 256;   // GRU hidden
static constexpr int kG   = 768;   // 3*H gates
static constexpr int kET  = 64;
static constexpr int kEA  = 128;
static constexpr int kCAT = 512;   // 2*H concat

// ---------------- bf16 helpers (storage-only, manual RNE convert) ----------
__device__ __forceinline__ __bf16 f2bf(float f) {
  union { float f; unsigned u; } a; a.f = f;
  unsigned r = a.u + 0x7FFFu + ((a.u >> 16) & 1u);
  union { unsigned short u; __bf16 b; } c; c.u = (unsigned short)(r >> 16);
  return c.b;
}
__device__ __forceinline__ float bf2f(__bf16 b) {
  union { unsigned short u; __bf16 b; } c; c.b = b;
  union { unsigned u; float f; } d; d.u = ((unsigned)c.u) << 16;
  return d.f;
}
__device__ __forceinline__ float sigf(float x) { return 1.f / (1.f + __expf(-x)); }

// ---------------- WMMA fragment loaders (ISA 7.12.2 layouts) ---------------
// A: 16x32 bf16 (MxK). lane 0-15: m=lane, K-halves 0-7/16-23; lanes 16-31: 8-15/24-31
__device__ __forceinline__ bf16x16 load_a_frag(const __bf16* base, int ld, int lane) {
  int m  = lane & 15;
  int kb = (lane & 16) ? 8 : 0;
  const __bf16* row = base + (long)m * ld + kb;
  bf16x16 f;
#pragma unroll
  for (int v = 0; v < 4; ++v) { f[2*v] = row[2*v];      f[2*v+1] = row[2*v+1]; }
#pragma unroll
  for (int v = 0; v < 4; ++v) { f[8+2*v] = row[16+2*v]; f[9+2*v] = row[16+2*v+1]; }
  return f;
}
// B: 32x16 bf16 (KxN), weights stored as W[N][K] row-major (= B^T).
// lanes 0-15: K=0-7/16-23, lanes 16-31: K=8-15/24-31; n = lane&15.
__device__ __forceinline__ bf16x16 load_b_frag(const __bf16* base, int ld, int lane) {
  int n  = lane & 15;
  int kb = (lane & 16) ? 8 : 0;
  const __bf16* row = base + (long)n * ld + kb;
  bf16x16 f;
#pragma unroll
  for (int e = 0; e < 16; ++e) f[e] = row[(e < 8) ? e : (e + 8)];
  return f;
}

// ---------------- f32 -> bf16 pack ----------------------------------------
__global__ void pack_bf16(const float* __restrict__ in, __bf16* __restrict__ out, long n) {
  long i = (long)blockIdx.x * blockDim.x + threadIdx.x;
  if (i < n) out[i] = f2bf(in[i]);
}

// ---------------- generic batched WMMA GEMM --------------------------------
// C[b][M,N] = act( A[b][M,K] (bf16, row-major, lda) * W[b][N,K]^T (bf16, ldw) + bias[N] )
// Each wave computes a 64x16 strip (4 M-tiles) so one weight fragment feeds
// 4 WMMAs; block = 8 waves. M must be a multiple of 64 (true for all calls).
__global__ __launch_bounds__(256)
void wmma_gemm(const __bf16* __restrict__ A, long strideA, int lda,
               const __bf16* __restrict__ W, long strideW, int ldw,
               const float* __restrict__ bias,
               void* __restrict__ Cout, long strideC, int ldc,
               int out_bf16, int relu, int M, int N, int K) {
  int lane = threadIdx.x & 31;
  int wave = threadIdx.x >> 5;
  int tile = blockIdx.x * 8 + wave;
  int ntn = N >> 4, ntm4 = M >> 6;
  if (tile >= ntm4 * ntn) return;         // wave-uniform: EXEC all-ones at WMMA
  int tg = tile / ntn, tn = tile % ntn;
  int batch = blockIdx.y;
  A += (long)batch * strideA;
  W += (long)batch * strideW;

  int n = tn * 16 + (lane & 15);
  float binit = bias ? bias[n] : 0.f;
  f32x8 acc[4];
#pragma unroll
  for (int i = 0; i < 4; ++i)
#pragma unroll
    for (int r = 0; r < 8; ++r) acc[i][r] = binit;

  const __bf16* Abase = A + (long)(tg * 64) * lda;
  const __bf16* Wbase = W + (long)(tn * 16) * ldw;
  for (int k0 = 0; k0 < K; k0 += 32) {
    bf16x16 wf = load_b_frag(Wbase + k0, ldw, lane);   // shared by 4 WMMAs
#pragma unroll
    for (int i = 0; i < 4; ++i) {
      bf16x16 af = load_a_frag(Abase + (long)(i * 16) * lda + k0, lda, lane);
      acc[i] = __builtin_amdgcn_wmma_f32_16x16x32_bf16(false, af, false, wf,
                                                       (short)0, acc[i], false, false);
    }
  }
  int mloc = (lane & 16) ? 8 : 0;
  if (out_bf16) {
    __bf16* C = (__bf16*)Cout + (long)batch * strideC;
#pragma unroll
    for (int i = 0; i < 4; ++i) {
      int mbase = tg * 64 + i * 16 + mloc;
#pragma unroll
      for (int r = 0; r < 8; ++r) {
        float v = acc[i][r]; if (relu) v = fmaxf(v, 0.f);
        C[(long)(mbase + r) * ldc + n] = f2bf(v);
      }
    }
  } else {
    float* C = (float*)Cout + (long)batch * strideC;
#pragma unroll
    for (int i = 0; i < 4; ++i) {
      int mbase = tg * 64 + i * 16 + mloc;
#pragma unroll
      for (int r = 0; r < 8; ++r) {
        float v = acc[i][r]; if (relu) v = fmaxf(v, 0.f);
        C[(long)(mbase + r) * ldc + n] = v;
      }
    }
  }
}

// ---------------- _pre MLP + (shift-add) + positional encoding -------------
// out_bf16[b,s,:] = sigmoid(sigmoid(x W1^T + b1) W2^T + b2) + PE(s)
__global__ __launch_bounds__(128)
void pre_kernel(const float* __restrict__ x,
                const float* __restrict__ W1, const float* __restrict__ b1,
                const float* __restrict__ W2, const float* __restrict__ b2,
                __bf16* __restrict__ out, int Sq, int E, float posscale, int do_shift) {
  __shared__ float xr[4][128];
  __shared__ float hr[4][32];
  int lane = threadIdx.x & 31, w = threadIdx.x >> 5;
  int row = blockIdx.x * 4 + w;
  int s = row % Sq;
  long rb = (long)row * E;
  for (int e = lane; e < E; e += 32) {
    float v = x[rb + e];
    if (do_shift && (s + 1 < Sq)) v += 0.55f * x[rb + E + e];  // ft[:, :-1] += .55*ft[:,1:]
    xr[w][e] = v;
  }
  __syncthreads();
  float acc = b1[lane];
  for (int e = 0; e < E; ++e) acc += xr[w][e] * W1[lane * E + e];
  hr[w][lane] = sigf(acc);
  __syncthreads();
  float neg = -logf(10000.f) / (float)E;
  for (int k = lane; k < E; k += 32) {
    float a2 = b2[k];
#pragma unroll
    for (int j = 0; j < 32; ++j) a2 += hr[w][j] * W2[k * 32 + j];
    float o = sigf(a2);
    float pos = (float)s * posscale;
    float dv = __expf((float)(k & ~1) * neg);
    float pe = (k & 1) ? __cosf(pos * dv) : __sinf(pos * dv);
    out[rb + k] = f2bf(o + pe);
  }
}

// ---------------- persistent GRU recurrence --------------------------------
// One block per direction (blockIdx.x: 0=fwd, 1=bwd reverse scan).
// 512 threads = 16 waves; per step: gh(16x768) = h_bf(16x256) @ Whh^T via WMMA,
// then fused GRU gate update; h kept in LDS as bf16. The 8 A-fragments of the
// step are preloaded once per wave and reused across its 3 N-tiles.
__global__ __launch_bounds__(512)
void gru_rec(const float* __restrict__ xp_f, const float* __restrict__ xp_b,
             const __bf16* __restrict__ Whh_f, const __bf16* __restrict__ Whh_b,
             const float* __restrict__ bhh_f, const float* __restrict__ bhh_b,
             __bf16* __restrict__ ycat, int Sq, int ldcat, int col_f, int col_b) {
  const int dir = blockIdx.x;
  const float*  xp  = dir ? xp_b  : xp_f;
  const __bf16* Whh = dir ? Whh_b : Whh_f;
  const float*  bhh = dir ? bhh_b : bhh_f;
  const int     col = dir ? col_b : col_f;
  const bool    rev = dir != 0;

  __shared__ __bf16 h_bf[kB * kH];   // 8 KB
  __shared__ float  gh[kB * kG];     // 48 KB

  int tid = threadIdx.x;
  int lane = tid & 31, wave = tid >> 5;
  for (int p = tid; p < kB * kH; p += 512) h_bf[p] = f2bf(0.f);
  __syncthreads();

  for (int st = 0; st < Sq; ++st) {
    int s = rev ? (Sq - 1 - st) : st;
    // ---- preload the step's 8 hidden-state fragments from LDS ------------
    bf16x16 afr[8];
#pragma unroll
    for (int kk = 0; kk < 8; ++kk)
      afr[kk] = load_a_frag((const __bf16*)h_bf + kk * 32, kH, lane);
    // ---- gh = h @ Whh^T + bhh : 48 N-tiles, 3 per wave -------------------
#pragma unroll
    for (int tt = 0; tt < 3; ++tt) {
      int n0 = (wave * 3 + tt) * 16;
      int n = n0 + (lane & 15);
      float binit = bhh[n];
      f32x8 acc;
#pragma unroll
      for (int r = 0; r < 8; ++r) acc[r] = binit;
#pragma unroll
      for (int kk = 0; kk < 8; ++kk) {
        bf16x16 wf = load_b_frag(Whh + (long)n0 * kH + kk * 32, kH, lane);
        acc = __builtin_amdgcn_wmma_f32_16x16x32_bf16(false, afr[kk], false, wf,
                                                      (short)0, acc, false, false);
      }
      int mb = (lane & 16) ? 8 : 0;
#pragma unroll
      for (int r = 0; r < 8; ++r) gh[(mb + r) * kG + n] = acc[r];
    }
    __syncthreads();
    // ---- fused gates + state update --------------------------------------
    for (int p = tid; p < kB * kH; p += 512) {
      int b = p >> 8, j = p & 255;
      long base = ((long)b * Sq + s) * kG;
      float xr_ = xp[base + j], xz_ = xp[base + kH + j], xn_ = xp[base + 2 * kH + j];
      float hr_ = gh[b * kG + j], hz_ = gh[b * kG + kH + j], hn_ = gh[b * kG + 2 * kH + j];
      float r  = sigf(xr_ + hr_);
      float z  = sigf(xz_ + hz_);
      float nn = tanhf(xn_ + r * hn_);
      float hold = bf2f(h_bf[p]);
      float hnew = (1.f - z) * nn + z * hold;
      h_bf[p] = f2bf(hnew);
      ycat[((long)b * Sq + s) * ldcat + col + j] = f2bf(hnew);
      if (st + 1 < Sq) {   // pull next step's xp rows into cache
        int s2 = rev ? (Sq - 2 - st) : (st + 1);
        __builtin_prefetch((const void*)&xp[((long)b * Sq + s2) * kG + j], 0, 0);
      }
    }
    __syncthreads();
  }
}

// ---------------- decoder scan: block reductions / scan (wave32) -----------
__device__ __forceinline__ float blockMax(float v, float* red, int lane, int wid) {
#pragma unroll
  for (int o = 16; o > 0; o >>= 1) v = fmaxf(v, __shfl_xor(v, o, 32));
  __syncthreads();
  if (lane == 0) red[wid] = v;
  __syncthreads();
  if (wid == 0) {
    float x = red[lane];
#pragma unroll
    for (int o = 16; o > 0; o >>= 1) x = fmaxf(x, __shfl_xor(x, o, 32));
    if (lane == 0) red[0] = x;
  }
  __syncthreads();
  return red[0];
}
__device__ __forceinline__ float blockSum(float v, float* red, int lane, int wid) {
#pragma unroll
  for (int o = 16; o > 0; o >>= 1) v += __shfl_xor(v, o, 32);
  __syncthreads();
  if (lane == 0) red[wid] = v;
  __syncthreads();
  if (wid == 0) {
    float x = red[lane];
#pragma unroll
    for (int o = 16; o > 0; o >>= 1) x += __shfl_xor(x, o, 32);
    if (lane == 0) red[0] = x;
  }
  __syncthreads();
  return red[0];
}
__device__ __forceinline__ float blockScanInc(float v, float* red, int lane, int wid) {
  float x = v;
#pragma unroll
  for (int o = 1; o < 32; o <<= 1) { float y = __shfl_up(x, (unsigned)o, 32); if (lane >= o) x += y; }
  __syncthreads();
  if (lane == 31) red[wid] = x;
  __syncthreads();
  if (wid == 0) {
    float w = red[lane];
#pragma unroll
    for (int o = 1; o < 32; o <<= 1) { float y = __shfl_up(w, (unsigned)o, 32); if (lane >= o) w += y; }
    red[lane] = w;
  }
  __syncthreads();
  float off = (wid > 0) ? red[wid - 1] : 0.f;
  __syncthreads();
  return x + off;
}

// scores pre-factored: score = wm*sA + wmb*sB (active) else sA+sB.
// grid = B, block = S = 1024 threads; carries m1/m2 live in registers (per-s).
__global__ __launch_bounds__(1024)
void decoder_kernel(const float* __restrict__ sA, const float* __restrict__ sB,
                    const unsigned char* __restrict__ mask_a,
                    const unsigned char* __restrict__ mask_t,
                    float* __restrict__ out, int T, int S) {
  int b = blockIdx.x;
  int s = threadIdx.x;
  int lane = s & 31, wid = s >> 5;
  __shared__ float red[32];
  bool ma = mask_a[b * S + s] != 0;
  float m1v = 0.f, m2v = 0.f;   // carries for this s

  for (int t = 0; t < T; ++t) {
    float sa = sA[((long)b * T + t) * S + s];
    float sb = sB[((long)b * T + t) * S + s];
    float wm = fminf(fmaxf(m2v, 0.f), 1.f);
    wm = (wm < 0.1f) ? 0.f : ((wm > 0.1f) ? 1.f : wm);
    float wmb = fminf(fmaxf(m1v, 0.f), 1.f);
    wmb = (wmb < 0.1f) ? 0.f : wmb;
    bool active = (t >= 2);
    float score = active ? (wm * sa + wmb * sb) : (sa + sb);
    bool ok = ma && (!active || (wm > 0.1f));
    score = ok ? score : -1.0e9f;

    float mx  = blockMax(score, red, lane, wid);
    float e   = __expf(score - mx);
    float den = blockSum(e, red, lane, wid);
    float qm  = mask_t[b * T + t] ? 1.f : 0.f;
    float wsl = e / den * qm;

    float gate = (t == 0) ? 1.f : (active ? wm : m1v);
    float osum = blockSum(wsl * gate * (float)s, red, lane, wid);
    if (s == 0 && t < T - 1) out[b * (T - 1) + t] = osum;

    float csum = blockScanInc(wsl, red, lane, wid);
    m2v = m1v;
    m1v = csum;
  }
}

// ---------------------------------------------------------------------------
// host launcher
// ---------------------------------------------------------------------------
enum {
  IN_FT = 0, IN_MT, IN_FA, IN_MA,
  TP_W1, TP_B1, TP_W2, TP_B2,
  AP_W1, AP_B1, AP_W2, AP_B2,
  T0F_WIH, T0F_WHH, T0F_BIH, T0F_BHH,
  T0B_WIH, T0B_WHH, T0B_BIH, T0B_BHH,
  T1F_WIH, T1F_WHH, T1F_BIH, T1F_BHH,
  T1B_WIH, T1B_WHH, T1B_BIH, T1B_BHH,
  T_WOUT, T_BOUT,
  A0F_WIH, A0F_WHH, A0F_BIH, A0F_BHH,
  A0B_WIH, A0B_WHH, A0B_BIH, A0B_BHH,
  A1F_WIH, A1F_WHH, A1F_BIH, A1F_BHH,
  A1B_WIH, A1B_WHH, A1B_BIH, A1B_BHH,
  A_WOUT, A_BOUT
};

extern "C" void kernel_launch(void* const* d_in, const int* in_sizes, int n_in,
                              void* d_out, int out_size, void* d_ws, size_t ws_size,
                              hipStream_t stream) {
  (void)in_sizes; (void)n_in; (void)out_size; (void)ws_size;
  char* ws = (char*)d_ws;
  size_t off = 0;
  auto alloc = [&](size_t bytes) -> void* {
    void* p = ws + off;
    off += (bytes + 255) & ~(size_t)255;
    return p;
  };
  auto fin = [&](int idx) { return (const float*)d_in[idx]; };
  auto packw = [&](int idx, long n) -> const __bf16* {
    void* dst = alloc((size_t)n * 2);
    pack_bf16<<<dim3((unsigned)((n + 255) / 256)), 256, 0, stream>>>(fin(idx), (__bf16*)dst, n);
    return (const __bf16*)dst;
  };

  // --- pack all weights to bf16 once --------------------------------------
  const __bf16* wT0F = packw(T0F_WIH, (long)kG * kET);
  const __bf16* uT0F = packw(T0F_WHH, (long)kG * kH);
  const __bf16* wT0B = packw(T0B_WIH, (long)kG * kET);
  const __bf16* uT0B = packw(T0B_WHH, (long)kG * kH);
  const __bf16* wT1F = packw(T1F_WIH, (long)kG * kCAT);
  const __bf16* uT1F = packw(T1F_WHH, (long)kG * kH);
  const __bf16* wT1B = packw(T1B_WIH, (long)kG * kCAT);
  const __bf16* uT1B = packw(T1B_WHH, (long)kG * kH);
  const __bf16* wToz = packw(T_WOUT, (long)kH * kCAT);
  const __bf16* wA0F = packw(A0F_WIH, (long)kG * kEA);
  const __bf16* uA0F = packw(A0F_WHH, (long)kG * kH);
  const __bf16* wA0B = packw(A0B_WIH, (long)kG * kEA);
  const __bf16* uA0B = packw(A0B_WHH, (long)kG * kH);
  const __bf16* wA1F = packw(A1F_WIH, (long)kG * kCAT);
  const __bf16* uA1F = packw(A1F_WHH, (long)kG * kH);
  const __bf16* wA1B = packw(A1B_WIH, (long)kG * kCAT);
  const __bf16* uA1B = packw(A1B_WHH, (long)kG * kH);
  const __bf16* wAoz = packw(A_WOUT, (long)kH * kCAT);

  // --- activation workspace ------------------------------------------------
  __bf16* xt_bf = (__bf16*)alloc(2L * kB * kT * kET);
  __bf16* xa_bf = (__bf16*)alloc(2L * kB * kS * kEA);
  float*  xpf   = (float*) alloc(4L * kB * kS * kG);   // reused per layer/encoder
  float*  xpb   = (float*) alloc(4L * kB * kS * kG);
  __bf16* catb  = (__bf16*)alloc(2L * kB * kS * kCAT); // reused h1cat/h2cat
  __bf16* et_bf = (__bf16*)alloc(2L * kB * kT * kH);
  __bf16* ea_bf = (__bf16*)alloc(2L * kB * kS * kH);
  float*  sAb   = (float*) alloc(4L * kB * kT * kS);
  float*  sBb   = (float*) alloc(4L * kB * kT * kS);

  // --- _pre + shift + positional encoding ----------------------------------
  pre_kernel<<<dim3(kB * kT / 4), 128, 0, stream>>>(
      fin(IN_FT), fin(TP_W1), fin(TP_B1), fin(TP_W2), fin(TP_B2),
      xt_bf, kT, kET, 8.344777745411855f, 1);
  pre_kernel<<<dim3(kB * kS / 4), 128, 0, stream>>>(
      fin(IN_FA), fin(AP_W1), fin(AP_B1), fin(AP_W2), fin(AP_B2),
      xa_bf, kS, kEA, 1.0f, 0);

  auto gemm = [&](const __bf16* A, long strA, int lda,
                  const __bf16* W, long strW, int ldw,
                  const float* bias, void* C, long strC, int ldc,
                  int M, int N, int K, int nb, int obf, int relu) {
    int tiles = (M / 64) * (N / 16);
    wmma_gemm<<<dim3((unsigned)((tiles + 7) / 8), (unsigned)nb), 256, 0, stream>>>(
        A, strA, lda, W, strW, ldw, bias, C, strC, ldc, obf, relu, M, N, K);
  };

  auto run_enc = [&](int Sq, int E, const __bf16* xin,
                     const __bf16* w0f, const __bf16* u0f, int b0fi, int b0fh,
                     const __bf16* w0b, const __bf16* u0b, int b0bi, int b0bh,
                     const __bf16* w1f, const __bf16* u1f, int b1fi, int b1fh,
                     const __bf16* w1b, const __bf16* u1b, int b1bi, int b1bh,
                     const __bf16* wout, int bouti, __bf16* eout) {
    int M = kB * Sq;
    // layer 0 input projections (big WMMA GEMMs)
    gemm(xin, 0, E, w0f, 0, E, fin(b0fi), xpf, 0, kG, M, kG, E, 1, 0, 0);
    gemm(xin, 0, E, w0b, 0, E, fin(b0bi), xpb, 0, kG, M, kG, E, 1, 0, 0);
    // layer 0 recurrence (fwd block + bwd block)
    gru_rec<<<dim3(2), 512, 0, stream>>>(xpf, xpb, u0f, u0b, fin(b0fh), fin(b0bh),
                                         catb, Sq, kCAT, 0, kH);
    // layer 1 projections from concat(h_f, h_b)
    gemm(catb, 0, kCAT, w1f, 0, kCAT, fin(b1fi), xpf, 0, kG, M, kG, kCAT, 1, 0, 0);
    gemm(catb, 0, kCAT, w1b, 0, kCAT, fin(b1bi), xpb, 0, kG, M, kG, kCAT, 1, 0, 0);
    gru_rec<<<dim3(2), 512, 0, stream>>>(xpf, xpb, u1f, u1b, fin(b1fh), fin(b1bh),
                                         catb, Sq, kCAT, 0, kH);
    // output projection + bias + relu, stored bf16 for score GEMMs
    gemm(catb, 0, kCAT, wout, 0, kCAT, fin(bouti), eout, 0, kH, M, kH, kCAT, 1, 1, 1);
  };

  run_enc(kT, kET, xt_bf,
          wT0F, uT0F, T0F_BIH, T0F_BHH, wT0B, uT0B, T0B_BIH, T0B_BHH,
          wT1F, uT1F, T1F_BIH, T1F_BHH, wT1B, uT1B, T1B_BIH, T1B_BHH,
          wToz, T_BOUT, et_bf);
  run_enc(kS, kEA, xa_bf,
          wA0F, uA0F, A0F_BIH, A0F_BHH, wA0B, uA0B, A0B_BIH, A0B_BHH,
          wA1F, uA1F, A1F_BIH, A1F_BHH, wA1B, uA1B, A1B_BIH, A1B_BHH,
          wAoz, A_BOUT, ea_bf);

  // --- pointer attention: factor scores into two batched Q*K^T GEMMs -------
  // sA[b,t,s] = <et[b,t,:128], ea[b,s,:128]>, sB over dims 128:256
  gemm(et_bf,       (long)kT * kH, kH, ea_bf,       (long)kS * kH, kH,
       nullptr, sAb, (long)kT * kS, kS, kT, kS, 128, kB, 0, 0);
  gemm(et_bf + 128, (long)kT * kH, kH, ea_bf + 128, (long)kS * kH, kH,
       nullptr, sBb, (long)kT * kS, kS, kT, kS, 128, kB, 0, 0);

  // --- sequential pointer scan (elementwise + block reductions) ------------
  decoder_kernel<<<dim3(kB), 1024, 0, stream>>>(
      sAb, sBb, (const unsigned char*)d_in[IN_MA], (const unsigned char*)d_in[IN_MT],
      (float*)d_out, kT, kS);
}
